// WaveletGuidedSSM_76098230550865
// MI455X (gfx1250) — matrix operand order repaired and checked by reading, not compiled, not run on hardware
//
#include <hip/hip_runtime.h>
#include <hip/hip_bf16.h>
#include <stdint.h>

typedef __attribute__((ext_vector_type(16))) _Float16 v16h;
typedef __attribute__((ext_vector_type(8)))  _Float16 v8h;
typedef __attribute__((ext_vector_type(8)))  float    v8f;
typedef __attribute__((ext_vector_type(4)))  unsigned int v4u;
typedef __attribute__((ext_vector_type(8)))  int      v8i;
typedef __attribute__((ext_vector_type(4)))  int      v4i;

#ifndef __has_builtin
#define __has_builtin(x) 0
#endif
#if defined(__AMDGCN__) && __has_builtin(__builtin_amdgcn_tensor_load_to_lds) && \
    __has_builtin(__builtin_amdgcn_s_wait_tensorcnt)
#define CDNA5_TDM 1
#else
#define CDNA5_TDM 0
#endif

// ---- problem dims (fixed by setup_inputs) ----
constexpr int Bn = 2, Hh = 128, Ww = 128, L = Hh * Ww;   // L = 16384
constexpr int DM = 128, DI = 256, DS = 16;
constexpr int OL = 66;                                    // wavelet out length

// filters = reversed DEC_LO / DEC_HI
__constant__ float c_flo[6] = { 0.3326705529509569f,  0.8068915093133388f,  0.4598775021193313f,
                               -0.13501102001039084f, -0.08544127388224149f, 0.035226291882100656f };
__constant__ float c_fhi[6] = { 0.035226291882100656f, 0.08544127388224149f, -0.13501102001039084f,
                               -0.4598775021193313f,   0.8068915093133388f, -0.3326705529509569f };

__device__ __forceinline__ float silu_f(float v) { return v / (1.f + __expf(-v)); }

#if CDNA5_TDM
// ---- Tensor Data Mover: flat 1-D copy global -> LDS (nelem f16 elements, nelem < 65536) ----
// D# packing per CDNA5 ISA sec 8.3/8.4 (group0: count/lds/global/type; group1: dims/strides).
__device__ __forceinline__ void tdm_copy_f16(const void* gsrc, unsigned lds_off_bytes, unsigned nelem) {
  unsigned long long ga = (unsigned long long)gsrc;
  v4u g0;
  g0[0] = 1u;                                            // count=1 valid descriptor
  g0[1] = lds_off_bytes;                                 // lds_addr
  g0[2] = (unsigned)(ga & 0xFFFFFFFFu);                  // global_addr[31:0]
  g0[3] = (unsigned)((ga >> 32) & 0x01FFFFFFu) | (2u << 30); // global_addr[56:32] | type=2
  v8i g1;
  g1[0] = 1 << 16;                                       // data_size = 1 (2 bytes/elem)
  g1[1] = (int)((nelem & 0xFFFFu) << 16);                // tensor_dim0[15:0]
  g1[2] = (int)((nelem >> 16) & 0xFFFFu) | (1 << 16);    // tensor_dim0[31:16] | tensor_dim1 = 1
  g1[3] = (int)((nelem & 0xFFFFu) << 16);                // tensor_dim1[31:16]=0 | tile_dim0 = nelem
  g1[4] = 1;                                             // tile_dim1 = 1, tile_dim2 = 0
  g1[5] = (int)nelem;                                    // tensor_dim0_stride[31:0]
  g1[6] = 0;                                             // stride0 hi | stride1 lo
  g1[7] = 0;
  v4i g2 = {0, 0, 0, 0};
  v4i g3 = {0, 0, 0, 0};
#if __clang_major__ >= 23
  v8i g4 = {0, 0, 0, 0, 0, 0, 0, 0};
  __builtin_amdgcn_tensor_load_to_lds(g0, g1, g2, g3, g4, 0);
#else
  __builtin_amdgcn_tensor_load_to_lds(g0, g1, g2, g3, 0);
#endif
}
#endif

// ---------------- k0: f32 -> f16 conversions + zero pooled accumulators ----------------
__global__ void k0_convert(const float* __restrict__ x, const float* __restrict__ win,
                           const float* __restrict__ wout,
                           _Float16* xh, _Float16* winh, _Float16* wouth, float* pool) {
  const int NX = Bn * L * DM, NW = 2 * DI * DM, NO = DM * DI;
  int i = blockIdx.x * 256 + threadIdx.x;
  if (i < NX) xh[i] = (_Float16)x[i];
  else if (i < NX + NW) winh[i - NX] = (_Float16)win[i - NX];
  else if (i < NX + NW + NO) wouth[i - NX - NW] = (_Float16)wout[i - NX - NW];
  if (i < 1024) pool[i] = 0.f;   // 3*256 pooled + modulation slots
}

// ---------------- k1a: row DWT (along w), coalesced in d ----------------
__global__ void k1a_rowdwt(const float* __restrict__ x, float* __restrict__ lo_g, float* __restrict__ hi_g) {
  extern __shared__ float s[];                 // [128 w][128 d]
  int b = blockIdx.x >> 7, h = blockIdx.x & 127;
  int tid = threadIdx.x;
  for (int idx = tid; idx < Hh * DM; idx += 256) {
    int w = idx >> 7, d = idx & 127;
    s[w * 128 + d] = x[((size_t)(b * L + h * Ww + w)) * DM + d];
  }
  __syncthreads();
  for (int idx = tid; idx < OL * DM; idx += 256) {
    int i = idx >> 7, d = idx & 127;
    float lo = 0.f, hi = 0.f;
#pragma unroll
    for (int j = 0; j < 6; ++j) {
      int w = 2 * i + j - 4;
      if (w >= 0 && w < Ww) { float v = s[w * 128 + d]; lo += c_flo[j] * v; hi += c_fhi[j] * v; }
    }
    size_t o = ((size_t)((b * Hh + h) * OL + i)) * DM + d;
    lo_g[o] = lo; hi_g[o] = hi;
  }
}

// ---------------- k1b: column DWT + pooled means (atomics) ----------------
__global__ void k1b_coldwt_pool(const float* __restrict__ lo_g, const float* __restrict__ hi_g,
                                float* __restrict__ pool) {
  extern __shared__ float s[];                 // lo[128*128], hi[128*128], red[256]
  float* slo = s; float* shi = s + Hh * DM; float* red = shi + Hh * DM;
  int b = blockIdx.x / OL, i = blockIdx.x % OL;
  int tid = threadIdx.x;
  for (int idx = tid; idx < Hh * DM; idx += 256) {
    int h = idx >> 7, d = idx & 127;
    size_t o = ((size_t)((b * Hh + h) * OL + i)) * DM + d;
    slo[idx] = lo_g[o]; shi[idx] = hi_g[o];
  }
  __syncthreads();
  float all = 0.f, ae = 0.f, ah = 0.f;
  for (int idx = tid; idx < OL * DM; idx += 256) {
    int io = idx >> 7, d = idx & 127;
    float ll = 0, lh = 0, hl = 0, hh = 0;
#pragma unroll
    for (int j = 0; j < 6; ++j) {
      int h = 2 * io + j - 4;
      if (h >= 0 && h < Hh) {
        float vl = slo[h * 128 + d], vh = shi[h * 128 + d];
        ll += c_flo[j] * vl; lh += c_fhi[j] * vl;
        hl += c_flo[j] * vh; hh += c_fhi[j] * vh;
      }
    }
    all += ll; ae += fabsf(lh) + fabsf(hl); ah += fabsf(hh);
  }
  const float sc = 1.f / ((float)OL * (float)OL);
  int d = tid & 127;
  red[tid] = all; __syncthreads();
  if (tid < 128) atomicAdd(&pool[0 * 256 + b * 128 + d], (red[tid] + red[tid + 128]) * sc);
  __syncthreads(); red[tid] = ae; __syncthreads();
  if (tid < 128) atomicAdd(&pool[1 * 256 + b * 128 + d], (red[tid] + red[tid + 128]) * sc);
  __syncthreads(); red[tid] = ah; __syncthreads();
  if (tid < 128) atomicAdd(&pool[2 * 256 + b * 128 + d], (red[tid] + red[tid + 128]) * sc);
}

// ---------------- k2: tiny modulation MLP ----------------
__global__ void k2_mod(const float* __restrict__ Wll, const float* __restrict__ Wedge,
                       const float* __restrict__ Whh, const float* __restrict__ WfB,
                       const float* __restrict__ WfC, const float* __restrict__ Wfdt,
                       float* __restrict__ pm) {
  __shared__ float ff[2][48];
  int tid = threadIdx.x;
  if (tid < 96) {
    int b = tid / 48, j = tid % 48;
    const float *w, *p;
    if (j < 16)      { w = Wll + j * DM;          p = pm + 0 * 256 + b * 128; }
    else if (j < 32) { w = Wedge + (j - 16) * DM; p = pm + 1 * 256 + b * 128; }
    else             { w = Whh + (j - 32) * DM;   p = pm + 2 * 256 + b * 128; }
    float a = 0.f;
    for (int d = 0; d < DM; ++d) a += w[d] * p[d];
    ff[b][j] = a;
  }
  __syncthreads();
  if (tid < 66) {
    int b = tid / 33, i = tid % 33;
    float a = 0.f;
    if (i < 16)      { for (int j = 0; j < 48; ++j) a += WfB[i * 48 + j] * ff[b][j]; pm[768 + b * 16 + i] = a; }
    else if (i < 32) { int ii = i - 16; for (int j = 0; j < 48; ++j) a += WfC[ii * 48 + j] * ff[b][j]; pm[800 + b * 16 + ii] = a; }
    else             { for (int j = 0; j < 48; ++j) a += Wfdt[j] * ff[b][j]; pm[832 + b] = a; }
  }
}

// ---------------- WMMA fragment helpers (f16 A/B, contiguous k order: element s of lane half h is k = 16h + s) ----------------
__device__ __forceinline__ v16h load_fragA(const _Float16* base) {
  // base points at row*K + kc + 16*half ; 16 contiguous K values
  v8h lo = *(const v8h*)(base);
  v8h hi = *(const v8h*)(base + 8);
  v16h a;
#pragma unroll
  for (int e = 0; e < 8; ++e) { a[e] = lo[e]; a[8 + e] = hi[e]; }
  return a;
}
__device__ __forceinline__ v16h load_fragB(const _Float16* base) {
  // base points at n*K + kc + 16*half ; 16 contiguous K values
  v8h lo = *(const v8h*)(base);
  v8h hi = *(const v8h*)(base + 8);
  v16h b;
#pragma unroll
  for (int e = 0; e < 8; ++e) { b[e] = lo[e]; b[8 + e] = hi[e]; }
  return b;
}

// ---------------- k3: xz = x @ W_in.T  (M=32768, N=512, K=128) ----------------
// TDM path: one wave DMAs the block's A (128x128 f16) and B (64x128 f16) tiles into LDS
// (TENSORcnt), all 8 waves then feed WMMA from LDS (ds_load_b128).
__global__ void k3_ingemm(const _Float16* __restrict__ xh, const _Float16* __restrict__ winh,
                          float* __restrict__ xz) {
  int tid = threadIdx.x, wave = tid >> 5, lane = tid & 31;
  int half = lane >> 4, ln = lane & 15;
  int m0 = blockIdx.x * 128 + wave * 16;
  int n0 = blockIdx.y * 64;
  v8f acc[4] = {};
#if CDNA5_TDM
  extern __shared__ _Float16 lds[];          // B tile [64][128] then A tile [128][128]
  _Float16* ldsB = lds;
  _Float16* ldsA = lds + 64 * DM;
  if (tid < 32) {
    tdm_copy_f16(winh + (size_t)n0 * DM, 0u, 64 * DM);
    tdm_copy_f16(xh + (size_t)(blockIdx.x * 128) * DM, (unsigned)(64 * DM * 2), 128 * DM);
    __builtin_amdgcn_s_wait_tensorcnt(0);
  }
  __syncthreads();
  int rowAl = wave * 16 + ln;                // A row local to block
#pragma unroll
  for (int kc = 0; kc < DM; kc += 32) {
    v16h A = load_fragA(ldsA + rowAl * DM + kc + half * 16);
    int kb = kc + half * 16;
#pragma unroll
    for (int nt = 0; nt < 4; ++nt) {
      int nl = nt * 16 + ln;
      v16h Bm = load_fragB(ldsB + nl * DM + kb);
      acc[nt] = __builtin_amdgcn_wmma_f32_16x16x32_f16(false, A, false, Bm, (short)0, acc[nt], false, false);
    }
  }
#else
  int rowA = m0 + ln;
#pragma unroll
  for (int kc = 0; kc < DM; kc += 32) {
    v16h A = load_fragA(xh + (size_t)rowA * DM + kc + half * 16);
    int kb = kc + half * 16;
#pragma unroll
    for (int nt = 0; nt < 4; ++nt) {
      int n = n0 + nt * 16 + ln;
      v16h Bm = load_fragB(winh + (size_t)n * DM + kb);
      acc[nt] = __builtin_amdgcn_wmma_f32_16x16x32_f16(false, A, false, Bm, (short)0, acc[nt], false, false);
    }
  }
#endif
#pragma unroll
  for (int nt = 0; nt < 4; ++nt)
#pragma unroll
    for (int v = 0; v < 8; ++v) {
      int row = m0 + v + half * 8;
      int col = n0 + nt * 16 + ln;
      xz[(size_t)row * (2 * DI) + col] = acc[nt][v];
    }
}

// ---------------- k4a: causal depthwise conv (K=4) + SiLU; write x_conv and t-major u ----------------
__global__ void k4a_conv(const float* __restrict__ xz, const float* __restrict__ convw,
                         const float* __restrict__ convb, float* __restrict__ xconv,
                         float* __restrict__ ut) {
  extern __shared__ float xc[];                // [64][257] padded
  int bid = blockIdx.x; int b = bid >> 8; int l0 = (bid & 255) * 64;
  int c = threadIdx.x;
  float w0 = convw[c * 4 + 0], w1 = convw[c * 4 + 1], w2 = convw[c * 4 + 2], w3 = convw[c * 4 + 3];
  float bias = convb[c];
  size_t base = (size_t)b * L;
  float p0 = (l0 >= 3) ? xz[(base + l0 - 3) * 512 + c] : 0.f;
  float p1 = (l0 >= 2) ? xz[(base + l0 - 2) * 512 + c] : 0.f;
  float p2 = (l0 >= 1) ? xz[(base + l0 - 1) * 512 + c] : 0.f;
  for (int r = 0; r < 64; ++r) {
    float cur = xz[(base + l0 + r) * 512 + c];
    float v = fmaf(w0, p0, fmaf(w1, p1, fmaf(w2, p2, fmaf(w3, cur, bias))));
    v = silu_f(v);
    xconv[(base + l0 + r) * DI + c] = v;
    xc[r * 257 + c] = v;
    p0 = p1; p1 = p2; p2 = cur;
  }
  __syncthreads();
  int tl = threadIdx.x & 63, cg = threadIdx.x >> 6;
  for (int q = 0; q < 64; ++q) {
    int c2 = q * 4 + cg;
    ut[((size_t)(b * DI + c2)) * L + l0 + tl] = xc[tl * 257 + c2];
  }
}

// ---------------- k4b: x_dbl = x_conv @ W_xproj.T ; dt/B_bar/Cm (t-major) ----------------
__global__ void k4b_xproj(const float* __restrict__ xconv, const float* __restrict__ Wxp,
                          const float* __restrict__ pm, const float* __restrict__ alpha_p,
                          const float* __restrict__ beta_p, const float* __restrict__ gamma_p,
                          float* __restrict__ dt_g, float* __restrict__ bbar, float* __restrict__ cmod) {
  extern __shared__ float s[];
  float* xc = s;                  // [64][257]
  float* xd = s + 64 * 257;       // [64][33]
  float* dtl = xd + 64 * 33;      // [64]
  int bid = blockIdx.x; int b = bid >> 8; int l0 = (bid & 255) * 64;
  int tid = threadIdx.x;
  size_t base = (size_t)b * L;
  for (int idx = tid; idx < 64 * DI; idx += 256) {
    int r = idx >> 8, c = idx & 255;
    xc[r * 257 + c] = xconv[(base + l0 + r) * DI + c];
  }
  __syncthreads();
  for (int idx = tid; idx < 64 * 33; idx += 256) {
    int r = idx / 33, j = idx % 33;
    const float* w = Wxp + j * DI;
    const float* xr = xc + r * 257;
    float a = 0.f;
    for (int cu = 0; cu < DI; ++cu) a += xr[cu] * w[cu];
    xd[r * 33 + j] = a;
  }
  __syncthreads();
  float alpha = alpha_p[0], beta = beta_p[0], gamma = gamma_p[0];
  if (tid < 64) {
    float v = xd[tid * 33 + 32] + gamma * pm[832 + b];
    float sp = (v > 20.f) ? v : log1pf(__expf(v));
    float dt = fminf(fmaxf(sp, 0.001f), 0.1f);
    dtl[tid] = dt;
    dt_g[b * L + l0 + tid] = dt;
  }
  __syncthreads();
  for (int idx = tid; idx < 64 * 16; idx += 256) {
    int r = idx & 63, si = idx >> 6;
    float dt = dtl[r];
    float Bm = xd[r * 33 + si] + alpha * pm[768 + b * 16 + si];
    float Bb = fminf(fmaxf(dt * Bm, -10.f), 10.f);
    float Cv = xd[r * 33 + 16 + si] + beta * pm[800 + b * 16 + si];
    size_t o = ((size_t)(b * DS + si)) * L + l0 + r;
    bbar[o] = Bb; cmod[o] = Cv;
  }
}

// ---------------- k5: selective scan; 8192 independent recurrences, shfl reduce over s ----------------
__global__ void k5_scan(const float* __restrict__ A_log, const float* __restrict__ dt_g,
                        const float* __restrict__ ut, const float* __restrict__ bbar,
                        const float* __restrict__ cmod, float* __restrict__ ys) {
  int tid = threadIdx.x, wave = tid >> 5, lane = tid & 31;
  int pair = (blockIdx.x * 8 + wave) * 2 + (lane >> 4);
  int b = pair >> 8, d = pair & 255;
  int s = lane & 15;
  float A = -__expf(A_log[d * DS + s]);
  const float4* dtp = (const float4*)(dt_g + (size_t)b * L);
  const float4* up  = (const float4*)(ut + ((size_t)(b * DI + d)) * L);
  const float4* bbp = (const float4*)(bbar + ((size_t)(b * DS + s)) * L);
  const float4* cmp = (const float4*)(cmod + ((size_t)(b * DS + s)) * L);
  float* yo = ys + (size_t)b * L * DI + d;
  float h = 0.f;
  for (int i = 0; i < L / 4; ++i) {
    float4 dv = dtp[i], uv = up[i], bv = bbp[i], cv = cmp[i];
    float dta[4] = { dv.x, dv.y, dv.z, dv.w };
    float ua[4]  = { uv.x, uv.y, uv.z, uv.w };
    float ba[4]  = { bv.x, bv.y, bv.z, bv.w };
    float ca[4]  = { cv.x, cv.y, cv.z, cv.w };
#pragma unroll
    for (int k = 0; k < 4; ++k) {
      float a = fminf(fmaxf(dta[k] * A, -20.f), 0.f);
      float dec = __expf(a);
      h = fmaf(dec, h, ua[k] * ba[k]);
      float p = h * ca[k];
      p += __shfl_xor(p, 1, 16);
      p += __shfl_xor(p, 2, 16);
      p += __shfl_xor(p, 4, 16);
      p += __shfl_xor(p, 8, 16);
      if (s == 0) yo[(size_t)(i * 4 + k) * DI] = p;
    }
  }
}

// ---------------- k6a: y = y*silu(z) + x_conv*D; LayerNorm; write f16 y_n ----------------
__global__ void k6a_combine_ln(const float* __restrict__ ys, const float* __restrict__ xz,
                               const float* __restrict__ xconv, const float* __restrict__ Dp,
                               const float* __restrict__ lnw, const float* __restrict__ lnb,
                               _Float16* __restrict__ ynh) {
  int tid = threadIdx.x, wave = tid >> 5, lane = tid & 31;
  int bid = blockIdx.x;
  int b = bid >> 9, t0 = (bid & 511) * 32;
  for (int i = 0; i < 4; ++i) {
    int t = t0 + wave * 4 + i;
    size_t rb = (size_t)b * L + t;
    float yv[8]; float sum = 0.f, ssq = 0.f;
#pragma unroll
    for (int ii = 0; ii < 8; ++ii) {
      int c = lane + ii * 32;
      float z = xz[rb * 512 + 256 + c];
      float y = ys[rb * 256 + c] * silu_f(z) + xconv[rb * 256 + c] * Dp[c];
      yv[ii] = y; sum += y; ssq += y * y;
    }
#pragma unroll
    for (int o = 16; o > 0; o >>= 1) { sum += __shfl_xor(sum, o, 32); ssq += __shfl_xor(ssq, o, 32); }
    float mu = sum * (1.f / 256.f);
    float var = ssq * (1.f / 256.f) - mu * mu;
    float rs = rsqrtf(var + 1e-5f);
#pragma unroll
    for (int ii = 0; ii < 8; ++ii) {
      int c = lane + ii * 32;
      ynh[rb * 256 + c] = (_Float16)((yv[ii] - mu) * rs * lnw[c] + lnb[c]);
    }
  }
}

// ---------------- k6b: out = x*skip + y_n @ W_out.T  (M=32768, N=128, K=256) ----------------
__global__ void k6b_outgemm(const _Float16* __restrict__ ynh, const _Float16* __restrict__ wouth,
                            const float* __restrict__ x, const float* __restrict__ skip,
                            float* __restrict__ out) {
  int tid = threadIdx.x, wave = tid >> 5, lane = tid & 31;
  int half = lane >> 4, ln = lane & 15;
  int m0 = blockIdx.x * 64 + (wave >> 1) * 16;
  int n0 = (wave & 1) * 64;
  int rowA = m0 + ln;
  v8f acc[4] = {};
#pragma unroll
  for (int kc = 0; kc < DI; kc += 32) {
    v16h A = load_fragA(ynh + (size_t)rowA * DI + kc + half * 16);
    int kb = kc + half * 16;
#pragma unroll
    for (int nt = 0; nt < 4; ++nt) {
      int n = n0 + nt * 16 + ln;
      v16h Bm = load_fragB(wouth + (size_t)n * DI + kb);
      acc[nt] = __builtin_amdgcn_wmma_f32_16x16x32_f16(false, A, false, Bm, (short)0, acc[nt], false, false);
    }
  }
#pragma unroll
  for (int nt = 0; nt < 4; ++nt)
#pragma unroll
    for (int v = 0; v < 8; ++v) {
      int row = m0 + v + half * 8;
      int col = n0 + nt * 16 + ln;
      out[(size_t)row * DM + col] = acc[nt][v] + x[(size_t)row * DM + col] * skip[col];
    }
}

// ================= host launcher =================
extern "C" void kernel_launch(void* const* d_in, const int* in_sizes, int n_in,
                              void* d_out, int out_size, void* d_ws, size_t ws_size,
                              hipStream_t stream) {
  (void)in_sizes; (void)n_in; (void)out_size; (void)ws_size;
  const float* x     = (const float*)d_in[0];
  const float* W_in  = (const float*)d_in[1];
  const float* convw = (const float*)d_in[2];
  const float* convb = (const float*)d_in[3];
  const float* W_ll  = (const float*)d_in[4];
  const float* W_edge= (const float*)d_in[5];
  const float* W_hh  = (const float*)d_in[6];
  const float* W_xp  = (const float*)d_in[7];
  const float* W_fB  = (const float*)d_in[8];
  const float* W_fC  = (const float*)d_in[9];
  const float* W_fdt = (const float*)d_in[10];
  const float* alpha = (const float*)d_in[11];
  const float* beta  = (const float*)d_in[12];
  const float* gamma = (const float*)d_in[13];
  const float* A_log = (const float*)d_in[14];
  const float* Dp    = (const float*)d_in[15];
  const float* skip  = (const float*)d_in[16];
  const float* W_out = (const float*)d_in[17];
  const float* lnw   = (const float*)d_in[18];
  const float* lnb   = (const float*)d_in[19];
  float* out = (float*)d_out;

  // workspace carve-out
  char* ws = (char*)d_ws;
  size_t off = 0;
  auto alloc = [&](size_t bytes) { size_t r = off; off = (off + bytes + 255) & ~(size_t)255; return r; };
  size_t oXH   = alloc((size_t)Bn * L * DM * 2);
  size_t oWINH = alloc((size_t)2 * DI * DM * 2);
  size_t oWOUT = alloc((size_t)DM * DI * 2);
  size_t oLO   = alloc((size_t)Bn * Hh * OL * DM * 4);
  size_t oHI   = alloc((size_t)Bn * Hh * OL * DM * 4);
  size_t oPOOL = alloc(1024 * 4);
  size_t oXZ   = alloc((size_t)Bn * L * 2 * DI * 4);
  size_t oXC   = alloc((size_t)Bn * L * DI * 4);
  size_t oUT   = alloc((size_t)Bn * DI * L * 4);
  size_t oDT   = alloc((size_t)Bn * L * 4);
  size_t oBB   = alloc((size_t)Bn * DS * L * 4);
  size_t oCM   = alloc((size_t)Bn * DS * L * 4);
  size_t oYS   = alloc((size_t)Bn * L * DI * 4);
  size_t oYNH  = alloc((size_t)Bn * L * DI * 2);

  _Float16* xh    = (_Float16*)(ws + oXH);
  _Float16* winh  = (_Float16*)(ws + oWINH);
  _Float16* wouth = (_Float16*)(ws + oWOUT);
  float* lo_g  = (float*)(ws + oLO);
  float* hi_g  = (float*)(ws + oHI);
  float* pool  = (float*)(ws + oPOOL);
  float* xz    = (float*)(ws + oXZ);
  float* xconv = (float*)(ws + oXC);
  float* ut    = (float*)(ws + oUT);
  float* dt_g  = (float*)(ws + oDT);
  float* bbar  = (float*)(ws + oBB);
  float* cmod  = (float*)(ws + oCM);
  float* ys    = (float*)(ws + oYS);
  _Float16* ynh = (_Float16*)(ws + oYNH);

  const int NCVT = Bn * L * DM + 2 * DI * DM + DM * DI;
  k0_convert<<<(NCVT + 255) / 256, 256, 0, stream>>>(x, W_in, W_out, xh, winh, wouth, pool);
  k1a_rowdwt<<<Bn * Hh, 256, Hh * DM * 4, stream>>>(x, lo_g, hi_g);
  k1b_coldwt_pool<<<Bn * OL, 256, (2 * Hh * DM + 256) * 4, stream>>>(lo_g, hi_g, pool);
  k2_mod<<<1, 128, 0, stream>>>(W_ll, W_edge, W_hh, W_fB, W_fC, W_fdt, pool);
  // 48 KB LDS for the TDM-staged A/B tiles (ignored by the fallback path).
  k3_ingemm<<<dim3((Bn * L) / 128, (2 * DI) / 64), 256, (64 + 128) * DM * 2, stream>>>(xh, winh, xz);
  k4a_conv<<<Bn * (L / 64), 256, 64 * 257 * 4, stream>>>(xz, convw, convb, xconv, ut);
  k4b_xproj<<<Bn * (L / 64), 256, (64 * 257 + 64 * 33 + 64) * 4, stream>>>(
      xconv, W_xp, pool, alpha, beta, gamma, dt_g, bbar, cmod);
  k5_scan<<<(Bn * DI) / 16, 256, 0, stream>>>(A_log, dt_g, ut, bbar, cmod, ys);
  k6a_combine_ln<<<Bn * (L / 32), 256, 0, stream>>>(ys, xz, xconv, Dp, lnw, lnb, ynh);
  k6b_outgemm<<<(Bn * L) / 64, 256, 0, stream>>>(ynh, wouth, x, skip, out);
}
